// MultiQueryAttention_81157702025388
// MI455X (gfx1250) — matrix operand order repaired and checked by reading, not compiled, not run on hardware
//
#include <hip/hip_runtime.h>
#include <hip/hip_bf16.h>

typedef __attribute__((ext_vector_type(16))) __bf16 v16bf;
typedef __attribute__((ext_vector_type(8)))  __bf16 v8bf;
typedef __attribute__((ext_vector_type(4)))  __bf16 v4bf;
typedef __attribute__((ext_vector_type(8)))  float  v8f;

#define BQ   2
#define TQ   2048
#define DIN  2048
#define NH   16
#define HD   128
#define DOUT 2048

#define BLK  128   // block tile M and N
#define BK   32    // k-step

static __device__ __forceinline__ v8f wmma_bf16(v16bf a, v16bf b, v8f c) {
    // D = A(16x32 bf16) x B(32x16 bf16) + C(16x16 f32)
    return __builtin_amdgcn_wmma_f32_16x16x32_bf16(
        false, a, false, b, (short)0, c, false, false);
}

// async 16-byte global -> LDS copy (ASYNCcnt-tracked, per-lane addresses)
static __device__ __forceinline__ void async_cp16(unsigned ldsoff, const __bf16* g) {
    asm volatile("global_load_async_to_lds_b128 %0, %1, off"
                 :: "v"(ldsoff), "v"(g) : "memory");
}
static __device__ __forceinline__ void wait_async0() {
    asm volatile("s_wait_asynccnt 0x0" ::: "memory");
}

// ---- fragment loaders -------------------------------------------------------
// A-frag (16x32, 16-bit): lane holds row M=lane&15; element j -> K = 16*half + j
static __device__ __forceinline__ v16bf ld_afrag_bf(const __bf16* rowp, int k0, int half) {
    const v8bf* p0 = (const v8bf*)(rowp + k0 + 16 * half);
    const v8bf* p1 = (const v8bf*)(rowp + k0 + 16 * half + 8);
    v8bf a = *p0, b = *p1;
    v16bf r;
#pragma unroll
    for (int j = 0; j < 8; ++j) { r[j] = a[j]; r[8 + j] = b[j]; }
    return r;
}
// B-frag (32x16, 16-bit): lane holds col N=lane&15; element j -> K = 16*half + j
static __device__ __forceinline__ v16bf ld_bfrag_bf(const __bf16* rowp, int k0, int half) {
    const v8bf* p = (const v8bf*)(rowp + k0 + 16 * half);
    v8bf a = p[0], b = p[1];
    v16bf r;
#pragma unroll
    for (int j = 0; j < 8; ++j) { r[j] = a[j]; r[8 + j] = b[j]; }
    return r;
}

// ---- fp32 -> bf16 pack (vectorized, count % 4 == 0) ------------------------
__global__ void __launch_bounds__(256) cvt_bf16_kernel(
    const float* __restrict__ in, __bf16* __restrict__ o, int nvec) {
    int i = blockIdx.x * 256 + threadIdx.x;
    if (i < nvec) {
        float4 v = ((const float4*)in)[i];
        v4bf r;
        r[0] = (__bf16)v.x; r[1] = (__bf16)v.y; r[2] = (__bf16)v.z; r[3] = (__bf16)v.w;
        ((v4bf*)o)[i] = r;
    }
}

// ---- GEMM: C[M,N] = A[M,K] * W[N,K]^T, bf16 in, fp32 out -------------------
// Block = 256 threads = 8 waves computes a 128x128 C tile. A/W k-slabs are
// double-buffered in LDS via async global->LDS copies; wave w computes rows
// 16w..16w+15 x 128 cols (8 WMMAs per k-step) from LDS fragments.
// Requires M%128==0, N%128==0, K%32==0.
__global__ void __launch_bounds__(256) gemm_bf16_wmma(
    const __bf16* __restrict__ A, const __bf16* __restrict__ W, float* __restrict__ C,
    int M, int N, int K) {
    __shared__ __align__(32) __bf16 sA[2][BLK * BK];   // 8KB per buffer
    __shared__ __align__(32) __bf16 sB[2][BLK * BK];

    int tid  = threadIdx.x;
    int lane = tid & 31;
    int wave = tid >> 5;
    int nBlocks = N >> 7;
    int mb = blockIdx.x / nBlocks;
    int nb = blockIdx.x - mb * nBlocks;
    int n = lane & 15, half = lane >> 4;

    // staging assignment: thread copies 32B = row (tid>>1), cols (tid&1)*16..+15
    int rcopy = tid >> 1;
    int ecopy = (tid & 1) * 16;
    const __bf16* gA = A + (size_t)(mb * BLK + rcopy) * K + ecopy;
    const __bf16* gW = W + (size_t)(nb * BLK + rcopy) * K + ecopy;
    unsigned ldsA[2] = { (unsigned)(uintptr_t)&sA[0][0] + (unsigned)tid * 32u,
                         (unsigned)(uintptr_t)&sA[1][0] + (unsigned)tid * 32u };
    unsigned ldsB[2] = { (unsigned)(uintptr_t)&sB[0][0] + (unsigned)tid * 32u,
                         (unsigned)(uintptr_t)&sB[1][0] + (unsigned)tid * 32u };

    auto stage = [&](int buf, int k0) {
        const __bf16* ga = gA + k0;
        const __bf16* gw = gW + k0;
        async_cp16(ldsA[buf],      ga);
        async_cp16(ldsA[buf] + 16, ga + 8);
        async_cp16(ldsB[buf],      gw);
        async_cp16(ldsB[buf] + 16, gw + 8);
    };

    v8f acc[8];
#pragma unroll
    for (int t = 0; t < 8; ++t) acc[t] = v8f{};

    stage(0, 0);
    int buf = 0;
    for (int k0 = 0; k0 < K; k0 += BK) {
        wait_async0();        // this wave's copies into `buf` done
        __syncthreads();      // => ALL waves' copies into `buf` done
        if (k0 + BK < K) stage(buf ^ 1, k0 + BK);   // overlap next slab

        const __bf16* tA = &sA[buf][0];
        const __bf16* tB = &sB[buf][0];
        v16bf af = ld_afrag_bf(tA + (wave * 16 + n) * BK, 0, half);
#pragma unroll
        for (int t = 0; t < 8; ++t) {
            v16bf bfr = ld_bfrag_bf(tB + (t * 16 + n) * BK, 0, half);
            acc[t] = wmma_bf16(af, bfr, acc[t]);
        }
        buf ^= 1;
        // next iteration's barrier orders this compute before `buf` is restaged
    }
#pragma unroll
    for (int t = 0; t < 8; ++t)
#pragma unroll
        for (int i = 0; i < 8; ++i)
            C[(size_t)(mb * BLK + wave * 16 + i + 8 * half) * N + nb * BLK + t * 16 + n]
                = acc[t][i];
}

// ---- RoPE + bf16 conversion -------------------------------------------------
// grid.x = B*T, grid.y = NH+2 (0..15: q heads w/ RoPE, 16: k w/ RoPE,
// 17: v convert, stored TRANSPOSED as vbT[B, HD, T] for fast P*V B-frags)
__global__ void __launch_bounds__(64) rope_cvt_kernel(
    const float* __restrict__ qf, const float* __restrict__ kf, const float* __restrict__ vf,
    const float* __restrict__ cosb, const float* __restrict__ sinb,
    __bf16* __restrict__ qb, __bf16* __restrict__ kb, __bf16* __restrict__ vbT) {
    int bt = blockIdx.x;
    int hh = blockIdx.y;
    int d  = threadIdx.x;                 // 0..63 (half of HD)
    const float* cs = cosb + (size_t)bt * HD;
    const float* sn = sinb + (size_t)bt * HD;
    if (hh < NH) {
        const float* q = qf + ((size_t)bt * NH + hh) * HD;
        __bf16* o = qb + ((size_t)bt * NH + hh) * HD;
        float q1 = q[d], q2 = q[d + 64];
        o[d]      = (__bf16)(q1 * cs[d]      - q2 * sn[d]);
        o[d + 64] = (__bf16)(q2 * cs[d + 64] + q1 * sn[d + 64]);
    } else if (hh == NH) {
        const float* k = kf + (size_t)bt * HD;
        __bf16* o = kb + (size_t)bt * HD;
        float k1 = k[d], k2 = k[d + 64];
        o[d]      = (__bf16)(k1 * cs[d]      - k2 * sn[d]);
        o[d + 64] = (__bf16)(k2 * cs[d + 64] + k1 * sn[d + 64]);
    } else {
        const float* v = vf + (size_t)bt * HD;
        int b = bt / TQ, t = bt - b * TQ;
        __bf16* o = vbT + (size_t)b * HD * TQ + t;
        o[(size_t)d * TQ]        = (__bf16)v[d];
        o[(size_t)(d + 64) * TQ] = (__bf16)v[d + 64];
    }
}

// ---- flash-style causal MQA attention --------------------------------------
// One wave = 16 query rows x full head (128). 8 waves/block, private LDS slice
// each. V is consumed transposed (vbT[B,HD,T]) so P*V B-frags are b128 loads.
// Output stored as bf16 (feeds the bf16 WMMA output projection).
__global__ void __launch_bounds__(256) mqa_flash_kernel(
    const __bf16* __restrict__ qb, const __bf16* __restrict__ kb,
    const __bf16* __restrict__ vbT, __bf16* __restrict__ ao, float scale) {
    __shared__ __bf16 plds[8][16 * 32];
    int lane = threadIdx.x & 31;
    int w = threadIdx.x >> 5;
    int gw = blockIdx.x * 8 + w;
    const int tTiles = TQ / 16;
    int bh = gw / tTiles;
    int tIdx = gw - bh * tTiles;
    int bb = bh / NH, h = bh - bb * NH;
    int q0 = tIdx * 16;
    int n = lane & 15, half = lane >> 4;

    // Q fragments (A-layout), RoPE'd bf16, rows q0..q0+15, K = head dim 128
    const __bf16* qrow = qb + (((size_t)(bb * TQ + q0 + n)) * NH + h) * HD;
    v16bf qfrag[4];
#pragma unroll
    for (int kk = 0; kk < 4; ++kk) qfrag[kk] = ld_afrag_bf(qrow, kk * 32, half);

    const __bf16* kbase = kb + (size_t)bb * TQ * HD;
    const __bf16* vtb   = vbT + (size_t)bb * HD * TQ;

    float m_[8], l_[8];
#pragma unroll
    for (int i = 0; i < 8; ++i) { m_[i] = -1e30f; l_[i] = 0.0f; }
    v8f acc[8];
#pragma unroll
    for (int i = 0; i < 8; ++i) acc[i] = v8f{};

    for (int kb0 = 0; kb0 <= q0; kb0 += 32) {
        // S = Q * K^T for 32 keys (two 16x16 D tiles); batch all K-frag loads
        const __bf16* krow0 = kbase + (size_t)(kb0 + n) * HD;
        const __bf16* krow1 = kbase + (size_t)(kb0 + 16 + n) * HD;
        v16bf kf0[4], kf1[4];
#pragma unroll
        for (int kk = 0; kk < 4; ++kk) {
            kf0[kk] = ld_bfrag_bf(krow0, kk * 32, half);
            kf1[kk] = ld_bfrag_bf(krow1, kk * 32, half);
        }
        v8f s0 = v8f{}, s1 = v8f{};
#pragma unroll
        for (int kk = 0; kk < 4; ++kk) s0 = wmma_bf16(qfrag[kk], kf0[kk], s0);
#pragma unroll
        for (int kk = 0; kk < 4; ++kk) s1 = wmma_bf16(qfrag[kk], kf1[kk], s1);

        // start V B-frag loads early (transposed V: contiguous keys per lane)
        v16bf bv[8];
#pragma unroll
        for (int nt = 0; nt < 8; ++nt)
            bv[nt] = ld_bfrag_bf(vtb + (size_t)(nt * 16 + n) * TQ, kb0, half);

        bool partial = (kb0 + 31 > q0);
        float rm[8];
#pragma unroll
        for (int i = 0; i < 8; ++i) {
            float a0 = s0[i] * scale, a1 = s1[i] * scale;
            if (partial) {
                int qrw = q0 + i + 8 * half;
                if (kb0 + n > qrw)      a0 = -1e30f;
                if (kb0 + 16 + n > qrw) a1 = -1e30f;
            }
            s0[i] = a0; s1[i] = a1;
            float r = fmaxf(a0, a1);
            r = fmaxf(r, __shfl_xor(r, 1, 32));
            r = fmaxf(r, __shfl_xor(r, 2, 32));
            r = fmaxf(r, __shfl_xor(r, 4, 32));
            r = fmaxf(r, __shfl_xor(r, 8, 32));   // halves stay independent
            rm[i] = r;
        }
        float alpha[8];
#pragma unroll
        for (int i = 0; i < 8; ++i) {
            float mn = fmaxf(m_[i], rm[i]);
            alpha[i] = __expf(m_[i] - mn);
            float p0 = __expf(s0[i] - mn);
            float p1 = __expf(s1[i] - mn);
            s0[i] = p0; s1[i] = p1;
            float r = p0 + p1;
            r += __shfl_xor(r, 1, 32);
            r += __shfl_xor(r, 2, 32);
            r += __shfl_xor(r, 4, 32);
            r += __shfl_xor(r, 8, 32);
            l_[i] = l_[i] * alpha[i] + r;
            m_[i] = mn;
        }
        // P: D-layout -> A-layout via per-wave LDS slice
        __bf16* pw = &plds[w][0];
#pragma unroll
        for (int i = 0; i < 8; ++i) {
            int mrow = i + 8 * half;
            pw[mrow * 32 + n]      = (__bf16)s0[i];
            pw[mrow * 32 + 16 + n] = (__bf16)s1[i];
        }
        asm volatile("s_wait_dscnt 0" ::: "memory");
        v16bf pa = ld_afrag_bf(&plds[w][(size_t)n * 32], 0, half);
        // O = O*alpha + P*V  (8 output tiles of 16 cols)
#pragma unroll
        for (int nt = 0; nt < 8; ++nt) {
            v8f a = acc[nt];
#pragma unroll
            for (int i = 0; i < 8; ++i) a[i] *= alpha[i];
            acc[nt] = wmma_bf16(pa, bv[nt], a);
        }
    }
    // normalize + store bf16 [B,T,H,HD] (input to output projection)
#pragma unroll
    for (int nt = 0; nt < 8; ++nt)
#pragma unroll
        for (int i = 0; i < 8; ++i) {
            int row = q0 + i + 8 * half;
            ao[(((size_t)(bb * TQ + row)) * NH + h) * HD + nt * 16 + n] =
                (__bf16)(acc[nt][i] / l_[i]);
        }
}

// ---- host launch ------------------------------------------------------------
extern "C" void kernel_launch(void* const* d_in, const int* in_sizes, int n_in,
                              void* d_out, int out_size, void* d_ws, size_t ws_size,
                              hipStream_t stream) {
    const float* x    = (const float*)d_in[0];
    const float* cosb = (const float*)d_in[1];
    const float* sinb = (const float*)d_in[2];
    const float* Wq   = (const float*)d_in[3];
    const float* Wk   = (const float*)d_in[4];
    const float* Wv   = (const float*)d_in[5];
    const float* Wo   = (const float*)d_in[6];
    float* out = (float*)d_out;

    char* ws = (char*)d_ws;
    const size_t BT = (size_t)BQ * TQ;                  // 4096
    size_t off = 0;
    __bf16* xb  = (__bf16*)(ws + off); off += BT * DIN * 2;            // 16.8 MB
    __bf16* wqb = (__bf16*)(ws + off); off += (size_t)DOUT * DIN * 2;  //  8.4 MB
    __bf16* wkb = (__bf16*)(ws + off); off += (size_t)HD * DIN * 2;
    __bf16* wvb = (__bf16*)(ws + off); off += (size_t)HD * DIN * 2;
    __bf16* wob = (__bf16*)(ws + off); off += (size_t)DOUT * DOUT * 2; //  8.4 MB
    float*  qf  = (float*)(ws + off);  off += BT * DOUT * 4;           // 33.6 MB
    float*  kf  = (float*)(ws + off);  off += BT * HD * 4;
    float*  vf  = (float*)(ws + off);  off += BT * HD * 4;
    __bf16* qbb = (__bf16*)(ws + off); off += BT * DOUT * 2;           // 16.8 MB
    __bf16* kbb = (__bf16*)(ws + off); off += BT * HD * 2;
    __bf16* vbT = (__bf16*)(ws + off); off += BT * HD * 2;             // transposed V
    __bf16* aob = (__bf16*)(ws + off); off += BT * DOUT * 2;           // 16.8 MB

    // one-time bf16 packs (removes all conversions from GEMM hot loops)
    auto cvt = [&](const float* src, __bf16* dst, size_t cnt) {
        int nvec = (int)(cnt / 4);
        cvt_bf16_kernel<<<dim3((nvec + 255) / 256), 256, 0, stream>>>(src, dst, nvec);
    };
    cvt(x,  xb,  BT * DIN);
    cvt(Wq, wqb, (size_t)DOUT * DIN);
    cvt(Wk, wkb, (size_t)HD * DIN);
    cvt(Wv, wvb, (size_t)HD * DIN);
    cvt(Wo, wob, (size_t)DOUT * DOUT);

    // Q/K/V projections (async-LDS double-buffered bf16 WMMA GEMM)
    gemm_bf16_wmma<<<dim3((BT / BLK) * (DOUT / BLK)), 256, 0, stream>>>(xb, wqb, qf, (int)BT, DOUT, DIN);
    gemm_bf16_wmma<<<dim3((BT / BLK) * (HD   / BLK)), 256, 0, stream>>>(xb, wkb, kf, (int)BT, HD,   DIN);
    gemm_bf16_wmma<<<dim3((BT / BLK) * (HD   / BLK)), 256, 0, stream>>>(xb, wvb, vf, (int)BT, HD,   DIN);

    // RoPE + bf16 pack (+ V transpose)
    rope_cvt_kernel<<<dim3((unsigned)BT, NH + 2), 64, 0, stream>>>(qf, kf, vf, cosb, sinb, qbb, kbb, vbT);

    // Flash causal MQA; reference scales by 1/sqrt(T) (token len), kept faithful
    const float scale = 0.022097086912079612f;          // 1/sqrt(2048)
    mqa_flash_kernel<<<dim3(BQ * NH * (TQ / 16) / 8), 256, 0, stream>>>(qbb, kbb, vbT, aob, scale);

    // Output projection (bf16 x bf16 -> fp32 out)
    gemm_bf16_wmma<<<dim3((BT / BLK) * (DOUT / BLK)), 256, 0, stream>>>(aob, wob, out, (int)BT, DOUT, DOUT);
}
